// SingleHeadSelfAttention_45569603010604
// MI455X (gfx1250) — matrix operand order repaired and checked by reading, not compiled, not run on hardware
//
#include <hip/hip_runtime.h>

#define BATCH 16
#define SEQ   2048
#define DIM   256

static constexpr float ATT_SCALE = 0.0625f; // 1/sqrt(256)

typedef __attribute__((ext_vector_type(16))) __bf16 v16bf;
typedef __attribute__((ext_vector_type(8)))  float  v8f;
typedef __attribute__((ext_vector_type(4)))  int    v4i;

#define AS1 __attribute__((address_space(1)))
#define AS3 __attribute__((address_space(3)))

union Frag { unsigned int u[8]; v16bf v; };

__device__ __forceinline__ unsigned short f2bf(float f) {
    unsigned int u = __float_as_uint(f);
    u += 0x7FFFu + ((u >> 16) & 1u);          // round-to-nearest-even
    return (unsigned short)(u >> 16);
}
__device__ __forceinline__ unsigned int pack2bf(float a, float b) {
    return (unsigned int)f2bf(a) | ((unsigned int)f2bf(b) << 16);
}

// ---- CDNA5 async global->LDS copy (ASYNCcnt path), 16 bytes per lane ------
__device__ __forceinline__ void async_copy16(unsigned int* ldsDst, const void* gSrc) {
#if __has_builtin(__builtin_amdgcn_global_load_async_to_lds_b128)
    // prototype (from clang diagnostic): (int4 AS1* src, int4 AS3* dst, Ii, Ii)
    __builtin_amdgcn_global_load_async_to_lds_b128(
        (AS1 v4i*)(size_t)gSrc,
        (AS3 v4i*)(unsigned int)(size_t)ldsDst,   // flat->LDS: low 32 bits
        0, 0);
#else
    // fallback: emit the raw VGLOBAL async op (vdst = LDS byte offset)
    asm volatile("global_load_async_to_lds_b128 %0, %1, off"
                 :: "v"((unsigned int)(size_t)ldsDst),
                    "v"((unsigned long long)(size_t)gSrc)
                 : "memory");
#endif
}
__device__ __forceinline__ void wait_async() {
#if __has_builtin(__builtin_amdgcn_s_wait_asynccnt)
    __builtin_amdgcn_s_wait_asynccnt(0);
#else
    asm volatile("s_wait_asynccnt 0x0" ::: "memory");
#endif
}

// ---------------------------------------------------------------------------
// Kernel 1: QKV projection.  Y[m,e] = sum_d x[m,d] * w_qkv[e,d]
// M = B*S = 32768, N = 768, K = 256.  Block tile 64x64, 4 waves.  fp32 inputs
// must be rounded to bf16 in VGPRs, so staging stays on the VALU path here.
// Q,K written row-major bf16; V written transposed [B][D][S] bf16.
// ---------------------------------------------------------------------------
__global__ __launch_bounds__(128) void qkv_proj_kernel(
    const float* __restrict__ x, const float* __restrict__ w,
    unsigned short* __restrict__ qb, unsigned short* __restrict__ kb,
    unsigned short* __restrict__ vtb) {
    __shared__ unsigned int Xs[64][17];   // 64 rows x 16 bf16-pairs (+pad)
    __shared__ unsigned int Ws[64][17];

    const int tid = threadIdx.x, wv = tid >> 5, lane = tid & 31;
    const int half = lane >> 4, l16 = lane & 15;
    const int mBase = blockIdx.x * 64, nBase = blockIdx.y * 64;

    v8f acc[4] = {};

    for (int kk = 0; kk < 8; ++kk) {      // K = 256 in steps of 32
#pragma unroll
        for (int i = 0; i < 8; ++i) {
            int t = tid + i * 128;        // 0..1023: 64 rows x 16 pairs
            int row = t >> 4, p = t & 15;
            const float* xs = &x[(size_t)(mBase + row) * DIM + kk * 32 + p * 2];
            Xs[row][p] = pack2bf(xs[0], xs[1]);
            const float* ws = &w[(size_t)(nBase + row) * DIM + kk * 32 + p * 2];
            Ws[row][p] = pack2bf(ws[0], ws[1]);
        }
        __syncthreads();

        Frag a;                           // A: 16x32, rows = this wave's 16 m
#pragma unroll
        for (int v = 0; v < 8; ++v)
            a.u[v] = Xs[wv * 16 + l16][(v >> 2) * 8 + half * 4 + (v & 3)];
#pragma unroll
        for (int nt = 0; nt < 4; ++nt) {
            Frag b;                       // B: 32x16, N = nt*16 + lane%16
#pragma unroll
            for (int v = 0; v < 8; ++v)
                b.u[v] = Ws[nt * 16 + l16][(v >> 2) * 8 + half * 4 + (v & 3)];
            acc[nt] = __builtin_amdgcn_wmma_f32_16x16x32_bf16(
                false, a.v, false, b.v, (short)0, acc[nt], false, false);
        }
        __syncthreads();
    }

    // C layout: lane l16 = N, row M = v + 8*half
#pragma unroll
    for (int nt = 0; nt < 4; ++nt) {
        int n = nBase + nt * 16 + l16;
#pragma unroll
        for (int v = 0; v < 8; ++v) {
            int m = mBase + wv * 16 + v + 8 * half;
            unsigned short h = f2bf(acc[nt][v]);
            if (n < 256) {
                qb[(size_t)m * DIM + n] = h;
            } else if (n < 512) {
                kb[(size_t)m * DIM + (n - 256)] = h;
            } else {
                int bb = m >> 11, s = m & 2047;
                vtb[((size_t)(bb << 8) + (n - 512)) * SEQ + s] = h;
            }
        }
    }
}

// ---------------------------------------------------------------------------
// Kernel 2: flash attention.  Block = 2 waves, 32 query rows; stream 32-key
// tiles.  Q/K/V^T tiles staged with async global->LDS (ASYNCcnt), bypassing
// the VGPR round-trip.  Online softmax over 16-lane halves matches the WMMA
// C-layout, so alpha rescaling is lane-local.  Row strides are 16B-aligned
// for B128 async LDS writes.
// ---------------------------------------------------------------------------
__global__ __launch_bounds__(64) void attn_kernel(
    const unsigned short* __restrict__ qb, const unsigned short* __restrict__ kb,
    const unsigned short* __restrict__ vtb, unsigned short* __restrict__ ctxb) {
    __shared__ __align__(16) unsigned int   Qs[32][132];   // 528B row stride
    __shared__ __align__(16) unsigned int   Ks[32][132];
    __shared__ __align__(16) unsigned int   Vts[256][20];  // 80B row stride
    __shared__ unsigned short Ps[2][16][36];               // per-wave P tile

    const int tid = threadIdx.x, wv = tid >> 5, lane = tid & 31;
    const int half = lane >> 4, l16 = lane & 15;
    const int bIdx = blockIdx.y;
    const int qBase = blockIdx.x * 32;
    const size_t tokBase = (size_t)bIdx * SEQ;

    // Stage Q tile once: 32 rows x 256 bf16 = 1024 x 16B, async
    {
        const unsigned short* qsrc = qb + (tokBase + qBase) * DIM;
#pragma unroll
        for (int i = 0; i < 16; ++i) {
            int idx = tid + i * 64;
            int row = idx >> 5, c4 = idx & 31;   // 32 x 16B per row
            async_copy16(&Qs[row][c4 * 4], qsrc + (size_t)row * DIM + c4 * 8);
        }
    }

    float vmax[8], vsum[8];
    v8f ctx[16] = {};
#pragma unroll
    for (int i = 0; i < 8; ++i) { vmax[i] = -1e30f; vsum[i] = 0.0f; }

    for (int kt = 0; kt < SEQ / 32; ++kt) {
        const int kBase = kt * 32;
        __syncthreads();   // previous iteration's LDS readers done

        // K tile: 32 keys x 256 bf16 (async)
        const unsigned short* ksrc = kb + (tokBase + kBase) * DIM;
#pragma unroll
        for (int i = 0; i < 16; ++i) {
            int idx = tid + i * 64;
            int row = idx >> 5, c4 = idx & 31;
            async_copy16(&Ks[row][c4 * 4], ksrc + (size_t)row * DIM + c4 * 8);
        }
        // V^T tile: 256 d x 32 keys (async; contiguous along S in vtb)
        const unsigned short* vbase = vtb + (size_t)bIdx * DIM * SEQ + kBase;
#pragma unroll
        for (int i = 0; i < 16; ++i) {
            int idx = tid + i * 64;              // 1024 x 16B
            int drow = idx >> 2, q4i = idx & 3;  // 4 x 16B per d-row
            async_copy16(&Vts[drow][q4i * 4], vbase + (size_t)drow * SEQ + q4i * 8);
        }
        if (kt + 1 < SEQ / 32)   // pull next K tile toward L2/L0
            __builtin_prefetch(kb + (tokBase + kBase + 32) * DIM + tid * 16, 0, 0);
        wait_async();
        __syncthreads();

        // ---- S = (Q K^T) * scale : 2 tiles of 16x16, K-loop over D ----
        v8f sc[2] = {};
#pragma unroll
        for (int kk = 0; kk < 8; ++kk) {
            Frag a;
#pragma unroll
            for (int v = 0; v < 8; ++v)
                a.u[v] = Qs[wv * 16 + l16][kk * 16 + (v >> 2) * 8 + half * 4 + (v & 3)];
#pragma unroll
            for (int nt = 0; nt < 2; ++nt) {
                Frag b;
#pragma unroll
                for (int v = 0; v < 8; ++v)
                    b.u[v] = Ks[nt * 16 + l16][kk * 16 + (v >> 2) * 8 + half * 4 + (v & 3)];
                sc[nt] = __builtin_amdgcn_wmma_f32_16x16x32_bf16(
                    false, a.v, false, b.v, (short)0, sc[nt], false, false);
            }
        }
#pragma unroll
        for (int nt = 0; nt < 2; ++nt)
#pragma unroll
            for (int v = 0; v < 8; ++v) sc[nt][v] *= ATT_SCALE;

        // ---- online softmax (per-row state replicated across 16 lanes) ----
        float alpha[8];
#pragma unroll
        for (int v = 0; v < 8; ++v) {
            float m = fmaxf(sc[0][v], sc[1][v]);
            m = fmaxf(m, __shfl_xor(m, 1, 32));
            m = fmaxf(m, __shfl_xor(m, 2, 32));
            m = fmaxf(m, __shfl_xor(m, 4, 32));
            m = fmaxf(m, __shfl_xor(m, 8, 32));
            float mn = fmaxf(vmax[v], m);
            alpha[v] = __expf(vmax[v] - mn);
            vmax[v] = mn;
            sc[0][v] = __expf(sc[0][v] - mn);
            sc[1][v] = __expf(sc[1][v] - mn);
            float ps = sc[0][v] + sc[1][v];
            ps += __shfl_xor(ps, 1, 32);
            ps += __shfl_xor(ps, 2, 32);
            ps += __shfl_xor(ps, 4, 32);
            ps += __shfl_xor(ps, 8, 32);
            vsum[v] = vsum[v] * alpha[v] + ps;
        }
#pragma unroll
        for (int ct = 0; ct < 16; ++ct)
#pragma unroll
            for (int v = 0; v < 8; ++v) ctx[ct][v] *= alpha[v];

        // ---- C-layout -> A-layout for P via per-wave LDS tile ----
#pragma unroll
        for (int nt = 0; nt < 2; ++nt)
#pragma unroll
            for (int v = 0; v < 8; ++v)
                Ps[wv][v + 8 * half][nt * 16 + l16] = f2bf(sc[nt][v]);
        // wave-local LDS RAW: DS pipeline is in-order per wave

        // ---- ctx += P @ V  (K-dim = 32 keys, single WMMA step) ----
        Frag pa;
#pragma unroll
        for (int v = 0; v < 8; ++v) {
            int pA = (v >> 2) * 8 + half * 4 + (v & 3);
            pa.u[v] = *(const unsigned int*)&Ps[wv][l16][2 * pA];
        }
#pragma unroll
        for (int ct = 0; ct < 16; ++ct) {
            Frag b;
#pragma unroll
            for (int v = 0; v < 8; ++v)
                b.u[v] = Vts[ct * 16 + l16][(v >> 2) * 8 + half * 4 + (v & 3)];
            ctx[ct] = __builtin_amdgcn_wmma_f32_16x16x32_bf16(
                false, pa.v, false, b.v, (short)0, ctx[ct], false, false);
        }
    }

    // ---- normalize and write bf16 ctx (row-major) ----
#pragma unroll
    for (int v = 0; v < 8; ++v) vsum[v] = 1.0f / vsum[v];
#pragma unroll
    for (int ct = 0; ct < 16; ++ct) {
        int dcol = ct * 16 + l16;
#pragma unroll
        for (int v = 0; v < 8; ++v) {
            int s = qBase + wv * 16 + v + 8 * half;
            ctxb[(tokBase + s) * DIM + dcol] = f2bf(ctx[ct][v] * vsum[v]);
        }
    }
}

// ---------------------------------------------------------------------------
// Kernel 3: output projection.  out[m,e] = sum_d ctx[m,d]*w_out[e,d] + b[e]
// M = 32768, N = 256, K = 256.  ctx is already bf16 -> async global->LDS
// staging; W_out is fp32 -> converted through VGPRs.
// ---------------------------------------------------------------------------
__global__ __launch_bounds__(128) void out_proj_kernel(
    const unsigned short* __restrict__ ctxb, const float* __restrict__ w,
    const float* __restrict__ bias, float* __restrict__ out) {
    __shared__ __align__(16) unsigned int Cs[64][20];   // 80B row stride
    __shared__ unsigned int Ws[64][17];

    const int tid = threadIdx.x, wv = tid >> 5, lane = tid & 31;
    const int half = lane >> 4, l16 = lane & 15;
    const int mBase = blockIdx.x * 64, nBase = blockIdx.y * 64;

    v8f acc[4] = {};

    for (int kk = 0; kk < 8; ++kk) {
        // ctx tile (bf16 already): 64 rows x 32 = 256 x 16B, async
#pragma unroll
        for (int i = 0; i < 2; ++i) {
            int t = tid + i * 128;         // 0..255
            int row = t >> 2, c4 = t & 3;  // 4 x 16B per row
            async_copy16(&Cs[row][c4 * 4],
                         ctxb + (size_t)(mBase + row) * DIM + kk * 32 + c4 * 8);
        }
        // W tile: fp32 -> bf16 pairs
#pragma unroll
        for (int i = 0; i < 8; ++i) {
            int t = tid + i * 128;
            int row = t >> 4, p = t & 15;
            const float* ws = &w[(size_t)(nBase + row) * DIM + kk * 32 + p * 2];
            Ws[row][p] = pack2bf(ws[0], ws[1]);
        }
        wait_async();
        __syncthreads();

        Frag a;
#pragma unroll
        for (int v = 0; v < 8; ++v)
            a.u[v] = Cs[wv * 16 + l16][(v >> 2) * 8 + half * 4 + (v & 3)];
#pragma unroll
        for (int nt = 0; nt < 4; ++nt) {
            Frag b;
#pragma unroll
            for (int v = 0; v < 8; ++v)
                b.u[v] = Ws[nt * 16 + l16][(v >> 2) * 8 + half * 4 + (v & 3)];
            acc[nt] = __builtin_amdgcn_wmma_f32_16x16x32_bf16(
                false, a.v, false, b.v, (short)0, acc[nt], false, false);
        }
        __syncthreads();
    }

#pragma unroll
    for (int nt = 0; nt < 4; ++nt) {
        int n = nBase + nt * 16 + l16;
        float bv = bias[n];
#pragma unroll
        for (int v = 0; v < 8; ++v) {
            int m = mBase + wv * 16 + v + 8 * half;
            out[(size_t)m * DIM + n] = acc[nt][v] + bv;
        }
    }
}

// ---------------------------------------------------------------------------
extern "C" void kernel_launch(void* const* d_in, const int* in_sizes, int n_in,
                              void* d_out, int out_size, void* d_ws, size_t ws_size,
                              hipStream_t stream) {
    const float* x     = (const float*)d_in[0];
    const float* w_qkv = (const float*)d_in[1];
    const float* w_out = (const float*)d_in[2];
    const float* b_out = (const float*)d_in[3];
    float* out = (float*)d_out;

    const size_t N = (size_t)BATCH * SEQ * DIM;   // 8,388,608 elements
    unsigned short* qb   = (unsigned short*)d_ws; // bf16 Q   [B][S][D]
    unsigned short* kb   = qb + N;                // bf16 K   [B][S][D]
    unsigned short* vtb  = kb + N;                // bf16 V^T [B][D][S]
    unsigned short* ctxb = vtb + N;               // bf16 ctx [B][S][D]

    qkv_proj_kernel<<<dim3(BATCH * SEQ / 64, 12), 128, 0, stream>>>(
        x, w_qkv, qb, kb, vtb);
    attn_kernel<<<dim3(SEQ / 32, BATCH), 64, 0, stream>>>(
        qb, kb, vtb, ctxb);
    out_proj_kernel<<<dim3(BATCH * SEQ / 64, 4), 128, 0, stream>>>(
        ctxb, w_out, b_out, out);
}
